// GAE_GraphConv_1314259992767
// MI455X (gfx1250) — matrix operand order repaired and checked by reading, not compiled, not run on hardware
//
#include <hip/hip_runtime.h>

// Problem constants (match reference)
#define NN    4096
#define EE    65536
#define OUTC  200
#define OUTP  224    // OUT padded to multiple of 32 (decoder GEMM K)
#define WTROW 208    // OUT padded to multiple of 16 (encoder GEMM N tiles)

typedef __attribute__((ext_vector_type(16))) __bf16        v16bf;
typedef __attribute__((ext_vector_type(8)))  float         v8f;
typedef __attribute__((ext_vector_type(4)))  unsigned int  u32x4;

union FragBF { v16bf v; u32x4 u[2]; };

__device__ __forceinline__ unsigned short f2bf(float f) {
  unsigned int u = __float_as_uint(f);
  unsigned int r = u + 0x7FFFu + ((u >> 16) & 1u);   // round-to-nearest-even
  return (unsigned short)(r >> 16);
}

// ---------------------------------------------------------------------------
// fp32 -> bf16 bulk convert (4 elems / thread)
// ---------------------------------------------------------------------------
__global__ void k_cvt_bf16(const float* __restrict__ in,
                           unsigned short* __restrict__ out, long n) {
  long i = ((long)blockIdx.x * blockDim.x + threadIdx.x) * 4;
  if (i + 3 >= n) return;
  float4 v = *(const float4*)(in + i);
  out[i + 0] = f2bf(v.x);
  out[i + 1] = f2bf(v.y);
  out[i + 2] = f2bf(v.z);
  out[i + 3] = f2bf(v.w);
}

// ---------------------------------------------------------------------------
// transpose+pad: out[r*C + c] = (r<vr && c<vc) ? W[c*ldW + r] : 0   (bf16 out)
// Produces K-contiguous B-operand storage for WMMA.
// ---------------------------------------------------------------------------
__global__ void k_transpose_pad(const float* __restrict__ W, int ldW,
                                unsigned short* __restrict__ out,
                                int R, int C, int vr, int vc) {
  long idx = (long)blockIdx.x * blockDim.x + threadIdx.x;
  if (idx >= (long)R * C) return;
  int r = (int)(idx / C);
  int c = (int)(idx - (long)r * C);
  float v = (r < vr && c < vc) ? W[(size_t)c * ldW + r] : 0.0f;
  out[idx] = f2bf(v);
}

__global__ void k_zero_f32(float* __restrict__ p, long n) {
  long i = (long)blockIdx.x * blockDim.x + threadIdx.x;
  if (i < n) p[i] = 0.0f;
}

// ---------------------------------------------------------------------------
// GEMM1: h = x @ W_rel, r = x @ W_root   (x bf16 [NN,NN], Wt* bf16 [WTROW,NN])
// One wave = 32x16 tile of BOTH outputs: 2 M-tiles x 2 weight matrices = 4
// accumulators; both B fragments are shared across the two A fragments
// (8 b128 loads -> 4 WMMAs per K-step).
// ---------------------------------------------------------------------------
__global__ void __launch_bounds__(256)
k_gemm1_wmma(const unsigned short* __restrict__ xb,
             const unsigned short* __restrict__ wtr,
             const unsigned short* __restrict__ wtq,
             float* __restrict__ h, float* __restrict__ r) {
  const int NT = WTROW / 16;                       // 13 column tiles
  const int MP = NN / 32;                          // 128 row-pair tiles
  int wid  = (int)((blockIdx.x * blockDim.x + threadIdx.x) >> 5);
  int lane = (int)(threadIdx.x & 31);
  if (wid >= MP * NT) return;                      // wave-uniform exit
  int mp = wid / NT, nt = wid - mp * NT;
  int m0 = mp * 32, n0 = nt * 16;
  int l16  = lane & 15;
  int half = lane >> 4;

  const unsigned short* a0Base = xb  + (size_t)(m0 + l16) * NN + half * 16;
  const unsigned short* a1Base = a0Base + (size_t)16 * NN;
  const unsigned short* b1Base = wtr + (size_t)(n0 + l16) * NN + half * 16;
  const unsigned short* b2Base = wtq + (size_t)(n0 + l16) * NN + half * 16;

  v8f accH0 = {}, accH1 = {}, accR0 = {}, accR1 = {};
  for (int k0 = 0; k0 < NN; k0 += 32) {
    FragBF a0, a1, b1, b2;
    a0.u[0] = *(const u32x4*)(a0Base + k0);        // K = half*16 + 0..7
    a0.u[1] = *(const u32x4*)(a0Base + k0 + 8);    // K = half*16 + 8..15
    a1.u[0] = *(const u32x4*)(a1Base + k0);
    a1.u[1] = *(const u32x4*)(a1Base + k0 + 8);
    b1.u[0] = *(const u32x4*)(b1Base + k0);        // K = half*16 + 0..7
    b1.u[1] = *(const u32x4*)(b1Base + k0 + 8);    // K = half*16 + 8..15
    b2.u[0] = *(const u32x4*)(b2Base + k0);
    b2.u[1] = *(const u32x4*)(b2Base + k0 + 8);
    if (k0 + 32 < NN) {
      __builtin_prefetch(a0Base + k0 + 32, 0, 1);
      __builtin_prefetch(a1Base + k0 + 32, 0, 1);
      __builtin_prefetch(b1Base + k0 + 32, 0, 1);
      __builtin_prefetch(b2Base + k0 + 32, 0, 1);
    }
    accH0 = __builtin_amdgcn_wmma_f32_16x16x32_bf16(false, a0.v, false, b1.v,
                                                    (short)0, accH0, false, false);
    accH1 = __builtin_amdgcn_wmma_f32_16x16x32_bf16(false, a1.v, false, b1.v,
                                                    (short)0, accH1, false, false);
    accR0 = __builtin_amdgcn_wmma_f32_16x16x32_bf16(false, a0.v, false, b2.v,
                                                    (short)0, accR0, false, false);
    accR1 = __builtin_amdgcn_wmma_f32_16x16x32_bf16(false, a1.v, false, b2.v,
                                                    (short)0, accR1, false, false);
  }

  int col = n0 + l16;
  if (col < OUTC) {
    #pragma unroll
    for (int rr = 0; rr < 8; ++rr) {
      int row0 = m0 + rr + 8 * half;
      int row1 = row0 + 16;
      h[(size_t)row0 * OUTC + col] = accH0[rr];
      h[(size_t)row1 * OUTC + col] = accH1[rr];
      r[(size_t)row0 * OUTC + col] = accR0[rr];
      r[(size_t)row1 * OUTC + col] = accR1[rr];
    }
  }
}

// ---------------------------------------------------------------------------
// Edge scatter: agg[dst] += h[src] * w    (fp32 atomics, stays in L2)
// ---------------------------------------------------------------------------
__global__ void k_scatter(const int* __restrict__ ei, const float* __restrict__ ew,
                          const float* __restrict__ h, float* __restrict__ agg) {
  long idx = (long)blockIdx.x * blockDim.x + threadIdx.x;
  if (idx >= (long)EE * OUTC) return;
  int e = (int)(idx / OUTC);
  int f = (int)(idx - (long)e * OUTC);
  int s = ei[e];
  int d = ei[EE + e];
  atomicAdd(&agg[(size_t)d * OUTC + f], h[(size_t)s * OUTC + f] * ew[e]);
}

// ---------------------------------------------------------------------------
// z = relu(agg + b_rel + r)  ->  bf16, row-padded to OUTP (zeros past OUTC)
// ---------------------------------------------------------------------------
__global__ void k_build_z(const float* __restrict__ agg, const float* __restrict__ r,
                          const float* __restrict__ brel,
                          unsigned short* __restrict__ zb) {
  long idx = (long)blockIdx.x * blockDim.x + threadIdx.x;
  if (idx >= (long)NN * OUTP) return;
  int row = (int)(idx / OUTP);
  int f   = (int)(idx - (long)row * OUTP);
  float v = 0.0f;
  if (f < OUTC) {
    v = agg[(size_t)row * OUTC + f] + brel[f] + r[(size_t)row * OUTC + f];
    v = fmaxf(v, 0.0f);
  }
  zb[idx] = f2bf(v);
}

// ---------------------------------------------------------------------------
// GEMM3: out = z @ W_dec + b_dec   (z bf16 [NN,OUTP], Wdt bf16 [NN,OUTP])
// Block (8 waves) computes a 128x64 output tile. Per K-step the block stages
// the 64x32 bf16 B chunk into LDS once (row stride padded to 40 elements for
// bank-conflict-free ds_load_b128), then each wave runs 4 WMMAs sharing one
// A fragment. Output write (64 MB) is the HBM-bound term; B L2 traffic /8.
// ---------------------------------------------------------------------------
#define BS_STRIDE 40   // padded LDS row stride (elements); 80 B, 16 B aligned

__global__ void __launch_bounds__(256)
k_gemm3_wmma(const unsigned short* __restrict__ zb,
             const unsigned short* __restrict__ wdt,
             const float* __restrict__ bdec,
             float* __restrict__ out) {
  __shared__ __align__(16) unsigned short Bs[64 * BS_STRIDE];  // 5 KB

  int tid  = (int)threadIdx.x;
  int wave = tid >> 5;
  int lane = tid & 31;
  int l16  = lane & 15;
  int half = lane >> 4;

  // grid: (NN/128) * (NN/64) = 32 * 64 = 2048 blocks, exact
  int bm = (int)blockIdx.x / (NN / 64);
  int bn = (int)blockIdx.x - bm * (NN / 64);
  int m0 = bm * 128 + wave * 16;
  int n0 = bn * 64;

  const unsigned short* aBase = zb + (size_t)(m0 + l16) * OUTP + half * 16;

  // cooperative B staging: thread t loads 16 B of row (n0 + t/4), segment t%4
  const unsigned short* bSrc = wdt + (size_t)(n0 + (tid >> 2)) * OUTP + (tid & 3) * 8;
  unsigned short*       bDst = &Bs[(tid >> 2) * BS_STRIDE + (tid & 3) * 8];

  v8f acc[4] = {};
  for (int k0 = 0; k0 < OUTP; k0 += 32) {
    *(u32x4*)bDst = *(const u32x4*)(bSrc + k0);
    __syncthreads();

    FragBF a;
    a.u[0] = *(const u32x4*)(aBase + k0);
    a.u[1] = *(const u32x4*)(aBase + k0 + 8);

    #pragma unroll
    for (int j = 0; j < 4; ++j) {
      FragBF b;
      const unsigned short* bp = &Bs[(j * 16 + l16) * BS_STRIDE + half * 16];
      b.u[0] = *(const u32x4*)(bp);       // K = half*16 + 0..7
      b.u[1] = *(const u32x4*)(bp + 8);   // K = half*16 + 8..15
      acc[j] = __builtin_amdgcn_wmma_f32_16x16x32_bf16(false, a.v, false, b.v,
                                                       (short)0, acc[j], false, false);
    }
    __syncthreads();
  }

  #pragma unroll
  for (int j = 0; j < 4; ++j) {
    int col = n0 + j * 16 + l16;
    float bias = bdec[col];
    #pragma unroll
    for (int rr = 0; rr < 8; ++rr) {
      int row = m0 + rr + 8 * half;
      out[(size_t)row * NN + col] = acc[j][rr] + bias;
    }
  }
}

// ---------------------------------------------------------------------------
// Launcher
// ---------------------------------------------------------------------------
extern "C" void kernel_launch(void* const* d_in, const int* in_sizes, int n_in,
                              void* d_out, int out_size, void* d_ws, size_t ws_size,
                              hipStream_t stream) {
  const float* x      = (const float*)d_in[0];
  const int*   ei     = (const int*)  d_in[1];
  const float* ew     = (const float*)d_in[2];
  const float* W_rel  = (const float*)d_in[3];
  const float* b_rel  = (const float*)d_in[4];
  const float* W_root = (const float*)d_in[5];
  const float* W_dec  = (const float*)d_in[6];
  const float* b_dec  = (const float*)d_in[7];
  float* out = (float*)d_out;

  char* ws = (char*)d_ws;
  size_t off = 0;
  auto carve = [&](size_t bytes) -> char* {
    char* p = ws + off;
    off = (off + bytes + 255) & ~(size_t)255;
    return p;
  };
  unsigned short* xb  = (unsigned short*)carve((size_t)NN * NN * 2);     // x bf16
  unsigned short* wtr = (unsigned short*)carve((size_t)WTROW * NN * 2);  // W_rel^T bf16
  unsigned short* wtq = (unsigned short*)carve((size_t)WTROW * NN * 2);  // W_root^T bf16
  unsigned short* wdt = (unsigned short*)carve((size_t)NN * OUTP * 2);   // W_dec^T bf16
  float*          h   = (float*)carve((size_t)NN * OUTC * 4);
  float*          rb  = (float*)carve((size_t)NN * OUTC * 4);
  float*          agg = (float*)carve((size_t)NN * OUTC * 4);
  unsigned short* zbf = (unsigned short*)carve((size_t)NN * OUTP * 2);

  const int BLK = 256;

  // 1) x fp32 -> bf16
  {
    long n = (long)NN * NN;
    int blocks = (int)((n / 4 + BLK - 1) / BLK);
    k_cvt_bf16<<<blocks, BLK, 0, stream>>>(x, xb, n);
  }
  // 2) weight transposes into K-contiguous bf16
  {
    long n = (long)WTROW * NN;
    int blocks = (int)((n + BLK - 1) / BLK);
    k_transpose_pad<<<blocks, BLK, 0, stream>>>(W_rel, OUTC, wtr, WTROW, NN, OUTC, NN);
    k_transpose_pad<<<blocks, BLK, 0, stream>>>(W_root, OUTC, wtq, WTROW, NN, OUTC, NN);
  }
  {
    long n = (long)NN * OUTP;
    int blocks = (int)((n + BLK - 1) / BLK);
    k_transpose_pad<<<blocks, BLK, 0, stream>>>(W_dec, NN, wdt, NN, OUTP, NN, OUTC);
  }
  // 3) zero the scatter accumulator
  {
    long n = (long)NN * OUTC;
    int blocks = (int)((n + BLK - 1) / BLK);
    k_zero_f32<<<blocks, BLK, 0, stream>>>(agg, n);
  }
  // 4) encoder GEMMs (fused h and root projections, 2x2 register blocking)
  {
    int waves  = (NN / 32) * (WTROW / 16);   // 128 * 13 = 1664
    int blocks = (waves * 32 + BLK - 1) / BLK;
    k_gemm1_wmma<<<blocks, BLK, 0, stream>>>(xb, wtr, wtq, h, rb);
  }
  // 5) edge scatter-add
  {
    long n = (long)EE * OUTC;
    int blocks = (int)((n + BLK - 1) / BLK);
    k_scatter<<<blocks, BLK, 0, stream>>>(ei, ew, h, agg);
  }
  // 6) z = relu(agg + b_rel + root) -> bf16 padded
  {
    long n = (long)NN * OUTP;
    int blocks = (int)((n + BLK - 1) / BLK);
    k_build_z<<<blocks, BLK, 0, stream>>>(agg, rb, b_rel, zbf);
  }
  // 7) decoder GEMM with bias, LDS-staged B
  {
    int blocks = (NN / 128) * (NN / 64);     // 2048, exact
    k_gemm3_wmma<<<blocks, BLK, 0, stream>>>(zbf, wdt, b_dec, out);
  }
}
